// Model_66374424593076
// MI455X (gfx1250) — matrix operand order repaired and checked by reading, not compiled, not run on hardware
//
#include <hip/hip_runtime.h>

typedef __attribute__((ext_vector_type(16))) _Float16 v16h;
typedef __attribute__((ext_vector_type(8)))  float    v8f;
typedef __attribute__((ext_vector_type(4)))  unsigned int u32x4;
typedef __attribute__((ext_vector_type(8)))  int i32x8;
typedef __attribute__((ext_vector_type(4)))  int i32x4;

#define BB   64
#define SS   512
#define EE   300
#define EP   320
#define HH   384
#define DD   768
#define G4   1536
#define BSZ  32768   // B*S
#define NHD  12

#if __has_builtin(__builtin_amdgcn_tensor_load_to_lds) && __has_builtin(__builtin_amdgcn_s_wait_tensorcnt)
#define HAVE_TDM 1
#else
#define HAVE_TDM 0
#endif

__device__ __forceinline__ float sigf(float x){ return 1.0f/(1.0f + __expf(-x)); }

#if HAVE_TDM
// Issue a TDM 2-D tile load: tile_h rows x tile_w f16 elements, row stride
// row_stride_elems, into LDS byte offset lds_byte_off of this workgroup.
__device__ __forceinline__ void tdm_load_tile_2d(unsigned lds_byte_off,
    const void* gaddr, unsigned tile_w, unsigned tile_h, unsigned row_stride_elems){
  unsigned long long ga = (unsigned long long)(size_t)gaddr;
  u32x4 g0;
  g0[0] = 1u;                                            // count=1, user mode
  g0[1] = lds_byte_off;                                  // lds_addr
  g0[2] = (unsigned)(ga & 0xFFFFFFFFu);                  // global_addr[95:64]
  g0[3] = (unsigned)((ga >> 32) & 0x1FFFFFFu) | (2u << 30); // addr msbs | type=2
  unsigned td0 = row_stride_elems;                       // tensor_dim0
  unsigned td1 = 0x40000000u;                            // tensor_dim1 (huge)
  i32x8 g1;
  g1[0] = (int)(1u << 16);                               // data_size=1 (2 bytes)
  g1[1] = (int)((td0 & 0xFFFFu) << 16);                  // tensor_dim0 lo
  g1[2] = (int)(((td0 >> 16) & 0xFFFFu) | ((td1 & 0xFFFFu) << 16));
  g1[3] = (int)(((td1 >> 16) & 0xFFFFu) | (tile_w << 16)); // tile_dim0
  g1[4] = (int)(tile_h & 0xFFFFu);                       // tile_dim1
  g1[5] = (int)row_stride_elems;                         // tensor_dim0_stride lo
  g1[6] = 0;
  g1[7] = 0;
  i32x4 z = {0,0,0,0};
#if defined(__clang_major__) && (__clang_major__ >= 23)
  i32x8 z8 = {0,0,0,0,0,0,0,0};
  __builtin_amdgcn_tensor_load_to_lds(g0, g1, z, z, z8, 0);
#else
  __builtin_amdgcn_tensor_load_to_lds(g0, g1, z, z, 0);
#endif
}
#endif

// ---------------------------------------------------------------------------
// Elementwise / prep kernels
// ---------------------------------------------------------------------------
__global__ void k_embed(const int* __restrict__ ids, const float* __restrict__ tab,
                        _Float16* __restrict__ emb, float* __restrict__ mask){
  size_t i = (size_t)blockIdx.x*blockDim.x + threadIdx.x;
  size_t total = (size_t)(BSZ + 8) * EP;
  if (i >= total) return;
  int col = (int)(i % EP);
  size_t row = i / EP;
  if (row >= BSZ){ emb[i] = (_Float16)0.0f; return; }
  int id = ids[row];
  emb[i] = (col < EE) ? (_Float16)tab[(size_t)id*EE + col] : (_Float16)0.0f;
  if (col == 0)
    mask[row] = (id != 0 && id != 2 && id != 10000 && id != 10001) ? 1.0f : 0.0f;
}

// WMMA B-fragment swizzle. Fragment f = kt*(N/16)+nt holds, per lane, 16
// consecutive K-values (k = kt*32 + (lane>=16?16:0) + e) at column
// n = nt*16 + (lane&15): one frag = 2x b128 per lane at load time.
// Source is W[N,K] f32 (transposed use: B[k][n] = W[n][k]); zero-pad K..Kpad.
__global__ void k_swz_T(const float* __restrict__ src, _Float16* __restrict__ dst,
                        int N, int K, int Kpad){
  int i = blockIdx.x*blockDim.x + threadIdx.x;
  if (i >= Kpad*N) return;
  int e    = i & 15;
  int lane = (i >> 4) & 31;
  int f    = i >> 9;
  int NT = N >> 4;
  int kt = f / NT, nt = f % NT;
  int k = kt*32 + ((lane >= 16) ? 16 : 0) + e;
  int n = nt*16 + (lane & 15);
  dst[i] = (k < K) ? (_Float16)src[(size_t)n*K + k] : (_Float16)0.0f;
}

// Same but source already [K,N] row-major (conv weights).
__global__ void k_swz_D(const float* __restrict__ src, _Float16* __restrict__ dst,
                        int N, int K, int Kpad){
  int i = blockIdx.x*blockDim.x + threadIdx.x;
  if (i >= Kpad*N) return;
  int e    = i & 15;
  int lane = (i >> 4) & 31;
  int f    = i >> 9;
  int NT = N >> 4;
  int kt = f / NT, nt = f % NT;
  int k = kt*32 + ((lane >= 16) ? 16 : 0) + e;
  int n = nt*16 + (lane & 15);
  dst[i] = (k < K) ? (_Float16)src[(size_t)k*N + n] : (_Float16)0.0f;
}

__global__ void k_tanh16(_Float16* buf, int n){
  int i = blockIdx.x*blockDim.x + threadIdx.x;
  if (i < n) buf[i] = (_Float16)tanhf((float)buf[i]);
}

__global__ void k_f32h(const float* __restrict__ s, _Float16* __restrict__ d, int n){
  int i = blockIdx.x*blockDim.x + threadIdx.x;
  if (i < n) d[i] = (_Float16)s[i];
}

// ---------------------------------------------------------------------------
// Core WMMA GEMM: C[M,N] = A[M,K](f16,lda) x Bsw (fragment-swizzled f16)
// (+bias1+bias2)(+res)(+beta*C). Block tile 64x128, 8 waves, 4 wmma/wave.
// A tile staged to LDS via the Tensor Data Mover (TDM) when available.
// M%64==0, N%128==0, K%32==0 assumed.
// ---------------------------------------------------------------------------
__global__ void __launch_bounds__(256) k_gemm(
    const _Float16* __restrict__ A, int lda,
    const _Float16* __restrict__ Bsw,
    const float* __restrict__ bias1, const float* __restrict__ bias2,
    const _Float16* __restrict__ res,
    float* __restrict__ Cf, _Float16* __restrict__ Ch,
    int M, int N, int K, int beta, int act)
{
  __shared__ _Float16 As[64][32];   // sole LDS object -> LDS byte offset 0
  const int tid  = threadIdx.x;
  const int lane = tid & 31, wid = tid >> 5;
  const int mi = wid >> 1, nq = (wid & 1) * 4;
  const int m0 = blockIdx.y * 64, n0 = blockIdx.x * 128;
  const int NT = N >> 4;

  const v8f vzero = {};
  v8f acc[4];
#pragma unroll
  for (int i = 0; i < 4; ++i) acc[i] = vzero;

  const int ar = tid >> 2;          // 0..63
  const int ac = (tid & 3) * 8;     // 0,8,16,24

  for (int k0 = 0; k0 < K; k0 += 32){
    const int kt = k0 >> 5;
#if HAVE_TDM
    if (wid == 0){
      tdm_load_tile_2d(0u, A + (size_t)m0*lda + k0, 32u, 64u, (unsigned)lda);
      __builtin_amdgcn_s_wait_tensorcnt(0);
    }
#else
    union { uint4 v; unsigned u[4]; } ta;
    ta.v = *(const uint4*)(A + (size_t)(m0 + ar)*lda + k0 + ac);
    unsigned* da = (unsigned*)&As[ar][ac];
#pragma unroll
    for (int i = 0; i < 4; ++i) da[i] = ta.u[i];
#endif
    if (k0 + 32 < K)
      __builtin_prefetch(Bsw + ((size_t)((kt + 1)*NT + (n0 >> 4) + nq)*32 + lane)*16, 0, 1);
    __syncthreads();

    // A fragment (16x32) from LDS: 16 contiguous k per lane (k = 16*half + e)
    v16h a;
    const int r  = mi*16 + (lane & 15);
    const int kb = (lane >= 16) ? 16 : 0;
#pragma unroll
    for (int j = 0; j < 8; ++j){
      int kk = 2*j + kb;
      a[2*j]   = As[r][kk];
      a[2*j+1] = As[r][kk+1];
    }
    // B fragments direct from swizzled global (2x global_load_b128 per frag)
#pragma unroll
    for (int i = 0; i < 4; ++i){
      int nt = (n0 >> 4) + nq + i;
      union { uint4 q[2]; v16h b; } ub;
      const uint4* pb = (const uint4*)(Bsw + ((size_t)(kt*NT + nt)*32 + lane)*16);
      ub.q[0] = pb[0]; ub.q[1] = pb[1];
      acc[i] = __builtin_amdgcn_wmma_f32_16x16x32_f16(false, a, false, ub.b,
                                                      (short)0, acc[i], false, false);
    }
    __syncthreads();
  }

  // epilogue (C 16x16 f32 layout: VGPR v -> row v / v+8 by lane half)
#pragma unroll
  for (int i = 0; i < 4; ++i){
    int col = n0 + (nq + i)*16 + (lane & 15);
    float bsum = 0.0f;
    if (bias1) bsum += bias1[col];
    if (bias2) bsum += bias2[col];
#pragma unroll
    for (int v = 0; v < 8; ++v){
      int row = m0 + mi*16 + v + ((lane >= 16) ? 8 : 0);
      size_t idx = (size_t)row*N + col;
      float x = acc[i][v] + bsum;
      if (beta) x += Cf[idx];
      if (res)  x += (float)res[idx];
      if (act == 1) x = tanhf(x);
      if (Cf) Cf[idx] = x;
      if (Ch) Ch[idx] = (_Float16)x;
    }
  }
}

// ---------------------------------------------------------------------------
// Persistent BiLSTM layer kernel. grid.x = 2 (dir 0 fwd, 1 bwd), 1024 threads.
// LDS: h state [64][384] f16 (48KB) + gate preacts [64][1536] f16 (192KB).
// Per step: g = h @ WhhT via WMMA (64x1536x384), then gate math, h/c update.
// Whh streamed from L2 in fragment-swizzled order (32B contiguous per lane).
// ---------------------------------------------------------------------------
__global__ void __launch_bounds__(1024) k_lstm_layer(
    const _Float16* __restrict__ preF, const _Float16* __restrict__ preB,
    const _Float16* __restrict__ WswF, const _Float16* __restrict__ WswB,
    _Float16* __restrict__ hout, float* __restrict__ cstate)
{
  extern __shared__ _Float16 smem[];
  _Float16* hS = smem;                 // 64*384
  _Float16* gS = smem + 64*HH;         // 64*1536
  const int dir = blockIdx.x;
  const _Float16* pre = dir ? preB : preF;
  const _Float16* Wsw = dir ? WswB : WswF;
  float* cs = cstate + (size_t)dir * (64*HH);

  const int tid  = threadIdx.x;
  const int lane = tid & 31, wid = tid >> 5;
  const int mi = wid & 3, ng = wid >> 2;      // 4 M-tiles x 8 N-groups
  for (int i = tid; i < 64*HH; i += 1024){ hS[i] = (_Float16)0.0f; cs[i] = 0.0f; }
  __syncthreads();

  const int r   = mi*16 + (lane & 15);
  const int kbA = (lane >= 16) ? 16 : 0;
  const v8f vzero = {};

  for (int t = 0; t < SS; ++t){
    const int te = dir ? (SS - 1 - t) : t;
    v8f acc[12];
#pragma unroll
    for (int i = 0; i < 12; ++i) acc[i] = vzero;

    for (int kt = 0; kt < 12; ++kt){
      v16h a;
#pragma unroll
      for (int j = 0; j < 8; ++j){
        int kk = kt*32 + 2*j + kbA;
        a[2*j]   = hS[r*HH + kk];
        a[2*j+1] = hS[r*HH + kk + 1];
      }
#pragma unroll
      for (int i = 0; i < 12; ++i){
        int nt = ng*12 + i;
        union { uint4 q[2]; v16h b; } ub;
        const uint4* p = (const uint4*)(Wsw + ((size_t)(kt*96 + nt)*32 + lane)*16);
        ub.q[0] = p[0]; ub.q[1] = p[1];
        acc[i] = __builtin_amdgcn_wmma_f32_16x16x32_f16(false, a, false, ub.b,
                                                        (short)0, acc[i], false, false);
      }
    }
    // spill gate preacts to LDS
#pragma unroll
    for (int i = 0; i < 12; ++i){
      int nt  = ng*12 + i;
      int col = nt*16 + (lane & 15);
#pragma unroll
      for (int v = 0; v < 8; ++v){
        int row = mi*16 + v + ((lane >= 16) ? 8 : 0);
        gS[row*G4 + col] = (_Float16)acc[i][v];
      }
    }
    __syncthreads();
    // elementwise LSTM cell
    for (int idx = tid; idx < 64*HH; idx += 1024){
      int m = idx / HH, hc = idx % HH;
      const _Float16* pr = pre + (size_t)(m*SS + te)*G4;
      float gi = (float)gS[m*G4 + hc]         + (float)pr[hc];
      float gf = (float)gS[m*G4 + HH + hc]    + (float)pr[HH + hc];
      float gg = (float)gS[m*G4 + 2*HH + hc]  + (float)pr[2*HH + hc];
      float go = (float)gS[m*G4 + 3*HH + hc]  + (float)pr[3*HH + hc];
      float c = cs[idx];
      c = sigf(gf)*c + sigf(gi)*tanhf(gg);
      float h = sigf(go)*tanhf(c);
      cs[idx] = c;
      hS[m*HH + hc] = (_Float16)h;
      hout[(size_t)(m*SS + te)*DD + dir*HH + hc] = (_Float16)h;
    }
    __syncthreads();
  }
}

// ---------------------------------------------------------------------------
// Attention small ops
// ---------------------------------------------------------------------------
__global__ void k_headscore(const _Float16* __restrict__ x, const float* __restrict__ W,
                            const float* __restrict__ bv, const float* __restrict__ mask,
                            float* __restrict__ out){
  int i = blockIdx.x*blockDim.x + threadIdx.x;   // [b,h,s]
  if (i >= BB*NHD*SS) return;
  int s = i & (SS-1);
  int h = (i >> 9) % NHD;
  int b = i / (NHD*SS);
  const _Float16* xp = x + (size_t)(b*SS + s)*DD;
  const float* wp = W + h*DD;
  float sum = 0.0f;
  for (int d = 0; d < DD; ++d) sum += (float)xp[d]*wp[d];
  out[i] = (sum + bv[h])*0.125f + (1.0f - mask[b*SS + s])*(-10000.0f);
}

__global__ void k_softmax512(float* buf){
  __shared__ float red[128];
  float* row = buf + (size_t)blockIdx.x*SS;
  int t = threadIdx.x;
  float m = -1e30f;
  for (int j = t; j < SS; j += 128) m = fmaxf(m, row[j]);
  red[t] = m; __syncthreads();
  for (int s = 64; s > 0; s >>= 1){ if (t < s) red[t] = fmaxf(red[t], red[t+s]); __syncthreads(); }
  m = red[0]; __syncthreads();
  float sum = 0.0f;
  for (int j = t; j < SS; j += 128){ float e = __expf(row[j]-m); row[j] = e; sum += e; }
  red[t] = sum; __syncthreads();
  for (int s = 64; s > 0; s >>= 1){ if (t < s) red[t] += red[t+s]; __syncthreads(); }
  float inv = 1.0f/red[0];
  for (int j = t; j < SS; j += 128) row[j] *= inv;
}

__global__ void k_pool(const float* __restrict__ w, const _Float16* __restrict__ x,
                       float* __restrict__ out){
  int i = blockIdx.x*blockDim.x + threadIdx.x;   // [b,d]
  if (i >= BB*DD) return;
  int d = i % DD, b = i / DD, h = d >> 6;
  const float* wp = w + (size_t)(b*NHD + h)*SS;
  const _Float16* xp = x + (size_t)b*SS*DD + d;
  float s = 0.0f;
  for (int t = 0; t < SS; ++t) s += wp[t]*(float)xp[(size_t)t*DD];
  out[i] = s;
}

__global__ void k_bmul(const _Float16* __restrict__ src, const float* __restrict__ pool,
                       _Float16* __restrict__ dst){
  size_t i = (size_t)blockIdx.x*blockDim.x + threadIdx.x;
  if (i >= (size_t)BSZ*DD) return;
  int d = (int)(i % DD);
  int b = (int)(i / ((size_t)SS*DD));
  dst[i] = (_Float16)((float)src[i]*pool[b*DD + d]);
}

__global__ void k_alpha(const _Float16* __restrict__ t1, const float* __restrict__ w2,
                        const float* __restrict__ b2, const float* __restrict__ mask,
                        float* __restrict__ alpha){
  int i = blockIdx.x*blockDim.x + threadIdx.x;   // [b*s]
  if (i >= BSZ) return;
  const _Float16* p = t1 + (size_t)i*DD;
  float s = 0.0f;
  for (int d = 0; d < DD; ++d) s += (float)p[d]*w2[d];
  alpha[i] = __expf(s + b2[0]) * mask[i];
}

__global__ void k_asum(const float* __restrict__ alpha, float* __restrict__ sums){
  __shared__ float red[256];
  int b = blockIdx.x, t = threadIdx.x;
  float s = 0.0f;
  for (int j = t; j < SS; j += 256) s += alpha[b*SS + j];
  red[t] = s; __syncthreads();
  for (int k = 128; k > 0; k >>= 1){ if (t < k) red[t] += red[t+k]; __syncthreads(); }
  if (t == 0) sums[b] = red[0] + 1e-8f;
}

__global__ void k_upper(const _Float16* __restrict__ wv, const float* __restrict__ alpha,
                        const float* __restrict__ sums, float* __restrict__ up){
  int i = blockIdx.x*blockDim.x + threadIdx.x;   // [b,d]
  if (i >= BB*DD) return;
  int d = i % DD, b = i / DD;
  float s = 0.0f;
  for (int t = 0; t < SS; ++t)
    s += (float)wv[(size_t)(b*SS + t)*DD + d]*alpha[b*SS + t];
  up[i] = s / sums[b];
}

__global__ void k_convpool(const float* __restrict__ y, int ksize,
                           float* __restrict__ out, int off){
  int i = blockIdx.x*blockDim.x + threadIdx.x;   // [b,f]
  if (i >= BB*256) return;
  int f = i % 256, b = i / 256;
  float m = 0.0f;
  int T = SS - ksize + 1;
  for (int t = 0; t < T; ++t) m = fmaxf(m, y[((size_t)b*SS + t)*256 + f]);
  out[b*DD + off + f] = m;   // relu(max) == max(0, max)
}

__global__ void k_feat(const float* __restrict__ up, const float* __restrict__ un,
                       _Float16* __restrict__ feat){
  int i = blockIdx.x*blockDim.x + threadIdx.x;
  if (i >= BB*2*DD) return;
  int j = i % (2*DD), b = i / (2*DD);
  feat[i] = (_Float16)((j < DD) ? up[b*DD + j] : un[b*DD + j - DD]);
}

__global__ void k_fc(const _Float16* __restrict__ feat, const float* __restrict__ W,
                     const float* __restrict__ bias, float* __restrict__ out){
  int i = blockIdx.x*blockDim.x + threadIdx.x;   // [b,c]
  if (i >= BB*10) return;
  int c = i % 10, b = i / 10;
  float s = bias[c];
  const _Float16* fp = feat + (size_t)b*2*DD;
  const float* wp = W + (size_t)c*2*DD;
  for (int d = 0; d < 2*DD; ++d) s += (float)fp[d]*wp[d];
  out[i] = s;
}

// ---------------------------------------------------------------------------
#define GRID1(n) dim3(((n) + 255)/256)

extern "C" void kernel_launch(void* const* d_in, const int* in_sizes, int n_in,
                              void* d_out, int out_size, void* d_ws, size_t ws_size,
                              hipStream_t stream) {
  (void)in_sizes; (void)n_in; (void)out_size; (void)ws_size;
  const int*   ids  = (const int*)d_in[0];
  const float* embT = (const float*)d_in[1];
  const float* Wih[4], *Whh[4], *bih[4], *bhh[4];
  for (int p = 0; p < 4; ++p){
    Wih[p] = (const float*)d_in[2 + 4*p + 0];
    Whh[p] = (const float*)d_in[2 + 4*p + 1];
    bih[p] = (const float*)d_in[2 + 4*p + 2];
    bhh[p] = (const float*)d_in[2 + 4*p + 3];
  }
  const float* Wq  = (const float*)d_in[18]; const float* bq  = (const float*)d_in[19];
  const float* Wqa = (const float*)d_in[20]; const float* bqa = (const float*)d_in[21];
  const float* Wk  = (const float*)d_in[22]; const float* bk  = (const float*)d_in[23];
  const float* Wka = (const float*)d_in[24]; const float* bka = (const float*)d_in[25];
  const float* Wt  = (const float*)d_in[26]; const float* bt  = (const float*)d_in[27];
  const float* Wf1 = (const float*)d_in[28]; const float* bf1 = (const float*)d_in[29];
  const float* Wf2 = (const float*)d_in[30]; const float* bf2 = (const float*)d_in[31];
  const float* cw[3] = {(const float*)d_in[32], (const float*)d_in[34], (const float*)d_in[36]};
  const float* cb[3] = {(const float*)d_in[33], (const float*)d_in[35], (const float*)d_in[37]};
  const float* Wu  = (const float*)d_in[38]; const float* bu  = (const float*)d_in[39];
  const float* Wfc = (const float*)d_in[40]; const float* bfc = (const float*)d_in[41];
  float* outp = (float*)d_out;

  // --- workspace layout ---
  char* base = (char*)d_ws;
  size_t off = 0;
  auto ALLOC = [&](size_t n)->char*{ char* p = base + off; off += (n + 255) & ~(size_t)255; return p; };
  _Float16* emb16 = (_Float16*)ALLOC((size_t)(BSZ + 8)*EP*2);
  float*    mask  = (float*)   ALLOC((size_t)BSZ*4);
  _Float16* preF  = (_Float16*)ALLOC((size_t)BSZ*G4*2);
  _Float16* preB  = (_Float16*)ALLOC((size_t)BSZ*G4*2);
  _Float16* hbuf  = (_Float16*)ALLOC((size_t)BSZ*DD*2);
  _Float16* q16   = (_Float16*)ALLOC((size_t)BSZ*DD*2);
  _Float16* k16   = (_Float16*)ALLOC((size_t)BSZ*DD*2);
  _Float16* wv16  = (_Float16*)ALLOC((size_t)BSZ*DD*2);
  float*    qs    = (float*)   ALLOC((size_t)BB*NHD*SS*4);
  float*    poolQ = (float*)   ALLOC((size_t)BB*DD*4);
  float*    poolK = (float*)   ALLOC((size_t)BB*DD*4);
  float*    alpha = (float*)   ALLOC((size_t)BSZ*4);
  float*    asum  = (float*)   ALLOC(256*4);
  float*    upper = (float*)   ALLOC((size_t)BB*DD*4);
  float*    underB= (float*)   ALLOC((size_t)BB*DD*4);
  float*    convO = (float*)   ALLOC((size_t)BB*DD*4);
  _Float16* convO16=(_Float16*)ALLOC((size_t)BB*DD*2);
  _Float16* feat16= (_Float16*)ALLOC((size_t)BB*2*DD*2);
  float*    cstate= (float*)   ALLOC((size_t)2*64*HH*4);
  _Float16* WqT   = (_Float16*)ALLOC((size_t)DD*DD*2);
  _Float16* WkT   = (_Float16*)ALLOC((size_t)DD*DD*2);
  _Float16* WtT   = (_Float16*)ALLOC((size_t)DD*DD*2);
  _Float16* Wf1T  = (_Float16*)ALLOC((size_t)DD*DD*2);
  _Float16* WuT   = (_Float16*)ALLOC((size_t)DD*DD*2);
  _Float16* WihT[4];
  WihT[0] = (_Float16*)ALLOC((size_t)EP*G4*2);
  WihT[1] = (_Float16*)ALLOC((size_t)EP*G4*2);
  WihT[2] = (_Float16*)ALLOC((size_t)DD*G4*2);
  WihT[3] = (_Float16*)ALLOC((size_t)DD*G4*2);
  _Float16* Wsw[4];
  for (int p = 0; p < 4; ++p) Wsw[p] = (_Float16*)ALLOC((size_t)HH*G4*2);
  _Float16* conv16 = (_Float16*)ALLOC((size_t)9*EP*256*2);
  // reused regions (pre buffers are free after the recurrences)
  _Float16* t1    = preF;                                     // [BS,768] f16
  float*    convY = (float*)((char*)preF + (size_t)BSZ*DD*2); // [BS,256] f32
  _Float16* wvp16 = preB;                                     // [BS,768] f16

  // --- weight prep: everything goes to WMMA-fragment order f16 ---
  k_swz_T<<<GRID1(DD*DD),256,0,stream>>>(Wq,  WqT,  DD, DD, DD);
  k_swz_T<<<GRID1(DD*DD),256,0,stream>>>(Wk,  WkT,  DD, DD, DD);
  k_swz_T<<<GRID1(DD*DD),256,0,stream>>>(Wt,  WtT,  DD, DD, DD);
  k_swz_T<<<GRID1(DD*DD),256,0,stream>>>(Wf1, Wf1T, DD, DD, DD);
  k_swz_T<<<GRID1(DD*DD),256,0,stream>>>(Wu,  WuT,  DD, DD, DD);
  k_swz_T<<<GRID1(EP*G4),256,0,stream>>>(Wih[0], WihT[0], G4, EE, EP);
  k_swz_T<<<GRID1(EP*G4),256,0,stream>>>(Wih[1], WihT[1], G4, EE, EP);
  k_swz_T<<<GRID1(DD*G4),256,0,stream>>>(Wih[2], WihT[2], G4, DD, DD);
  k_swz_T<<<GRID1(DD*G4),256,0,stream>>>(Wih[3], WihT[3], G4, DD, DD);
  for (int p = 0; p < 4; ++p)
    k_swz_T<<<GRID1(HH*G4),256,0,stream>>>(Whh[p], Wsw[p], G4, HH, HH);
  {
    int ksz[3] = {2,3,4}; int sl = 0;
    for (int c = 0; c < 3; ++c)
      for (int dk = 0; dk < ksz[c]; ++dk, ++sl)
        k_swz_D<<<GRID1(EP*256),256,0,stream>>>(cw[c] + (size_t)dk*EE*256,
                                                conv16 + (size_t)sl*EP*256, 256, EE, EP);
  }

  // --- embedding + mask ---
  k_embed<<<GRID1((BSZ+8)*EP),256,0,stream>>>(ids, embT, emb16, mask);

  // --- BiLSTM layer 0: input projections then recurrence ---
  k_gemm<<<dim3(G4/128, BSZ/64),256,0,stream>>>(emb16, EP, WihT[0], bih[0], bhh[0],
          nullptr, nullptr, preF, BSZ, G4, EP, 0, 0);
  k_gemm<<<dim3(G4/128, BSZ/64),256,0,stream>>>(emb16, EP, WihT[1], bih[1], bhh[1],
          nullptr, nullptr, preB, BSZ, G4, EP, 0, 0);
  size_t lstm_lds = (size_t)(64*HH + 64*G4)*2;   // 240 KB of the WGP's 320 KB
  k_lstm_layer<<<2, 1024, lstm_lds, stream>>>(preF, preB, Wsw[0], Wsw[1], hbuf, cstate);

  // --- BiLSTM layer 1 ---
  k_gemm<<<dim3(G4/128, BSZ/64),256,0,stream>>>(hbuf, DD, WihT[2], bih[2], bhh[2],
          nullptr, nullptr, preF, BSZ, G4, DD, 0, 0);
  k_gemm<<<dim3(G4/128, BSZ/64),256,0,stream>>>(hbuf, DD, WihT[3], bih[3], bhh[3],
          nullptr, nullptr, preB, BSZ, G4, DD, 0, 0);
  k_lstm_layer<<<2, 1024, lstm_lds, stream>>>(preF, preB, Wsw[2], Wsw[3], hbuf, cstate);
  k_tanh16<<<GRID1(BSZ*DD),256,0,stream>>>(hbuf, BSZ*DD);   // hid = tanh(h1)

  // --- Fastformer-style additive attention ---
  k_gemm<<<dim3(DD/128, BSZ/64),256,0,stream>>>(hbuf, DD, WqT, bq, nullptr,
          nullptr, nullptr, q16, BSZ, DD, DD, 0, 0);
  k_gemm<<<dim3(DD/128, BSZ/64),256,0,stream>>>(hbuf, DD, WkT, bk, nullptr,
          nullptr, nullptr, k16, BSZ, DD, DD, 0, 0);
  k_headscore<<<GRID1(BB*NHD*SS),256,0,stream>>>(q16, Wqa, bqa, mask, qs);
  k_softmax512<<<BB*NHD, 128, 0, stream>>>(qs);
  k_pool<<<GRID1(BB*DD),256,0,stream>>>(qs, q16, poolQ);
  k_bmul<<<GRID1(BSZ*DD),256,0,stream>>>(k16, poolQ, k16);            // qk = k * pooled_q
  k_headscore<<<GRID1(BB*NHD*SS),256,0,stream>>>(k16, Wka, bka, mask, qs);
  k_softmax512<<<BB*NHD, 128, 0, stream>>>(qs);
  k_pool<<<GRID1(BB*DD),256,0,stream>>>(qs, k16, poolK);
  k_bmul<<<GRID1(BSZ*DD),256,0,stream>>>(q16, poolK, wvp16);          // pooled_k * q_heads
  k_gemm<<<dim3(DD/128, BSZ/64),256,0,stream>>>(wvp16, DD, WtT, bt, nullptr,
          q16, nullptr, wv16, BSZ, DD, DD, 0, 0);                     // wv = .@Wt + bt + q
  k_gemm<<<dim3(DD/128, BSZ/64),256,0,stream>>>(wv16, DD, Wf1T, bf1, nullptr,
          nullptr, nullptr, t1, BSZ, DD, DD, 0, 1);                   // tanh(wv@Wf1+bf1)
  k_alpha<<<GRID1(BSZ),256,0,stream>>>(t1, Wf2, bf2, mask, alpha);
  k_asum<<<BB, 256, 0, stream>>>(alpha, asum);
  k_upper<<<GRID1(BB*DD),256,0,stream>>>(wv16, alpha, asum, upper);

  // --- TextCNN path: K-accumulating WMMA GEMMs + relu-maxpool ---
  {
    int ksz[3] = {2,3,4}; int sl = 0;
    for (int c = 0; c < 3; ++c){
      for (int dk = 0; dk < ksz[c]; ++dk, ++sl)
        k_gemm<<<dim3(256/128, BSZ/64),256,0,stream>>>(emb16 + (size_t)dk*EP, EP,
                conv16 + (size_t)sl*EP*256,
                (dk == 0) ? cb[c] : nullptr, nullptr, nullptr,
                convY, nullptr, BSZ, 256, EP, (dk > 0) ? 1 : 0, 0);
      k_convpool<<<GRID1(BB*256),256,0,stream>>>(convY, ksz[c], convO, c*256);
    }
  }
  k_f32h<<<GRID1(BB*DD),256,0,stream>>>(convO, convO16, BB*DD);
  k_gemm<<<dim3(DD/128, 1),256,0,stream>>>(convO16, DD, WuT, bu, nullptr,
          nullptr, underB, nullptr, BB, DD, DD, 0, 0);

  // --- head ---
  k_feat<<<GRID1(BB*2*DD),256,0,stream>>>(upper, underB, feat16);
  k_fc<<<GRID1(BB*10),256,0,stream>>>(feat16, Wfc, bfc, outp);
}
